// SimCLRModel_75488345195250
// MI455X (gfx1250) — matrix operand order repaired and checked by reading, not compiled, not run on hardware
//
#include <hip/hip_runtime.h>

// ---------------------------------------------------------------------------
// SimCLR-GNN pipeline for MI455X (gfx1250, wave32, WMMA).
//   conv1 (WMMA, fused gather->MLP->scatter-max) -> BN1 -> conv2 (WMMA) -> BN2
//   -> sorted-batch mean/max pooling -> projector MLP (tiny, scalar).
// ---------------------------------------------------------------------------

#define NN       100000      // nodes
#define NE       1000000     // edges
#define NG       64          // graphs
#define HID      64
#define PROJ     128
#define MSTR     136         // 128 + 8 f16 pad (16B-aligned row stride)
#define W2STR    72          // 64 + 8 f16 pad
#define BNEPS    1e-5f

typedef __attribute__((ext_vector_type(8)))  _Float16 v8h;
typedef __attribute__((ext_vector_type(16))) _Float16 v16h;
typedef __attribute__((ext_vector_type(8)))  float    v8f;

__device__ __forceinline__ v16h cat8(v8h lo, v8h hi) {
  return __builtin_shufflevector(lo, hi, 0,1,2,3,4,5,6,7,8,9,10,11,12,13,14,15);
}

// float atomic max handling negatives (int for >=0, uint-min for <0)
__device__ __forceinline__ void atomicMaxF(float* addr, float v) {
  if (v >= 0.f) atomicMax((int*)addr, __float_as_int(v));
  else          atomicMin((unsigned int*)addr, __float_as_uint(v));
}

// ---------------------------------------------------------------------------
// Fused EdgeConv: out[d,c] = max over edges e with dst[e]==d of
//   relu(W2^T relu(W1^T [x_i ; x_j - x_i] + b1) + b2),  out pre-zeroed (>=0).
// 256 threads = 8 waves; 128 edges per tile; 24 v_wmma per tile.
// ---------------------------------------------------------------------------
__global__ __launch_bounds__(256) void edgeconv_wmma_kernel(
    const float* __restrict__ xin,
    const int* __restrict__ src, const int* __restrict__ dst,
    const float* __restrict__ w1, const float* __restrict__ b1,
    const float* __restrict__ w2, const float* __restrict__ b2,
    float* __restrict__ out)
{
  __shared__ _Float16 sM[128][MSTR];     // messages; cols 64..127 reused as hidden stage
  __shared__ _Float16 sW1[64][MSTR];     // W1 transposed: [outCol][K=0..127]
  __shared__ _Float16 sW2[64][W2STR];    // W2 transposed: [outCol][K=0..63]
  __shared__ float sB1[64], sB2[64];
  __shared__ int sDst[128];

  const int tid  = threadIdx.x;
  const int lane = tid & 31;
  const int wv   = tid >> 5;
  const int mrow = lane & 15;
  const int hi16 = lane >> 4;            // 0: lanes 0-15, 1: lanes 16-31

  // stage weights transposed + converted to f16 (once per block)
  for (int i = tid; i < 64 * 128; i += 256) {
    int n = i & 63, k = i >> 6;
    sW1[n][k] = (_Float16)w1[k * 64 + n];
  }
  for (int i = tid; i < 64 * 64; i += 256) {
    int n = i & 63, k = i >> 6;
    sW2[n][k] = (_Float16)w2[k * 64 + n];
  }
  if (tid < 64) { sB1[tid] = b1[tid]; sB2[tid] = b2[tid]; }
  __syncthreads();

  const int nTiles = (NE + 127) >> 7;
  for (int tile = blockIdx.x; tile < nTiles; tile += gridDim.x) {
    const int ebase = tile << 7;

    // ---- build message rows: 2 threads per edge, 32 channels each ----
    {
      const int er = tid >> 1;
      const int hf = tid & 1;
      const int e  = ebase + er;
      const int c0 = hf << 5;
      if (e < NE) {
        const int d = dst[e];
        const int s = src[e];
        if (hf == 0) sDst[er] = d;
        const float4* pi = (const float4*)(xin + (size_t)d * 64 + c0);
        const float4* pj = (const float4*)(xin + (size_t)s * 64 + c0);
        #pragma unroll
        for (int q = 0; q < 8; ++q) {
          float4 a = pi[q], bb = pj[q];
          const int c = c0 + q * 4;
          sM[er][c + 0] = (_Float16)a.x;  sM[er][64 + c + 0] = (_Float16)(bb.x - a.x);
          sM[er][c + 1] = (_Float16)a.y;  sM[er][64 + c + 1] = (_Float16)(bb.y - a.y);
          sM[er][c + 2] = (_Float16)a.z;  sM[er][64 + c + 2] = (_Float16)(bb.z - a.z);
          sM[er][c + 3] = (_Float16)a.w;  sM[er][64 + c + 3] = (_Float16)(bb.w - a.w);
        }
      } else {
        if (hf == 0) sDst[er] = -1;
        #pragma unroll
        for (int q = 0; q < 8; ++q) {
          const int c = c0 + q * 4;
          #pragma unroll
          for (int j = 0; j < 4; ++j) {
            sM[er][c + j] = (_Float16)0.f;
            sM[er][64 + c + j] = (_Float16)0.f;
          }
        }
      }
      // prefetch next tile's edge indices (global_prefetch_b8, speculative)
      const int enext = ebase + (int)(gridDim.x << 7) + er;
      if (enext < NE) {
        __builtin_prefetch(&dst[enext], 0, 1);
        __builtin_prefetch(&src[enext], 0, 1);
      }
    }
    __syncthreads();

    const int arow = (wv << 4) + mrow;

    // ---- A fragments for GEMM1 (16x32 f16 layout, K=0..127) ----
    v16h afr[4];
    #pragma unroll
    for (int kb = 0; kb < 4; ++kb) {
      const int k0 = kb * 32 + hi16 * 8;
      v8h lo = *(const v8h*)&sM[arow][k0];
      v8h hi = *(const v8h*)&sM[arow][k0 + 16];
      afr[kb] = cat8(lo, hi);
    }

    // ---- GEMM1: [16x128]@[128x64] + bias + relu -> stage f16 into sM[.. ][64..127]
    //      (safe: this wave's A frags already live in registers; rows are wave-private)
    #pragma unroll
    for (int nt = 0; nt < 4; ++nt) {
      v8f acc = {0.f,0.f,0.f,0.f,0.f,0.f,0.f,0.f};
      const int col = nt * 16 + mrow;
      #pragma unroll
      for (int kb = 0; kb < 4; ++kb) {
        const int kw = kb * 32 + hi16 * 8;
        v8h lo = *(const v8h*)&sW1[col][kw];
        v8h hi = *(const v8h*)&sW1[col][kw + 16];
        v16h bfr = cat8(lo, hi);
        acc = __builtin_amdgcn_wmma_f32_16x16x32_f16(
            false, afr[kb], false, bfr, (short)0, acc, false, false);
      }
      const float bias = sB1[col];
      #pragma unroll
      for (int r = 0; r < 8; ++r) {
        float v = acc[r] + bias; v = v > 0.f ? v : 0.f;   // C/D layout: M = r + hi16*8
        const int row = (wv << 4) + r + hi16 * 8;
        sM[row][64 + col] = (_Float16)v;
      }
    }
    __syncthreads();

    // ---- GEMM2: [16x64]@[64x64] + bias + relu -> atomic max scatter ----
    v16h afr2[2];
    #pragma unroll
    for (int kb = 0; kb < 2; ++kb) {
      const int k0 = kb * 32 + hi16 * 8;
      v8h lo = *(const v8h*)&sM[arow][64 + k0];
      v8h hi = *(const v8h*)&sM[arow][64 + k0 + 16];
      afr2[kb] = cat8(lo, hi);
    }
    #pragma unroll
    for (int nt = 0; nt < 4; ++nt) {
      v8f acc = {0.f,0.f,0.f,0.f,0.f,0.f,0.f,0.f};
      const int col = nt * 16 + mrow;
      #pragma unroll
      for (int kb = 0; kb < 2; ++kb) {
        const int kw = kb * 32 + hi16 * 8;
        v8h lo = *(const v8h*)&sW2[col][kw];
        v8h hi = *(const v8h*)&sW2[col][kw + 16];
        v16h bfr = cat8(lo, hi);
        acc = __builtin_amdgcn_wmma_f32_16x16x32_f16(
            false, afr2[kb], false, bfr, (short)0, acc, false, false);
      }
      const float bias = sB2[col];
      #pragma unroll
      for (int r = 0; r < 8; ++r) {
        const int d = sDst[(wv << 4) + r + hi16 * 8];
        if (d >= 0) {
          float v = acc[r] + bias; v = v > 0.f ? v : 0.f;
          // values >= 0 -> uint bit order == numeric order; out pre-zeroed
          atomicMax((unsigned int*)&out[(size_t)d * 64 + col], __float_as_uint(v));
        }
      }
    }
    __syncthreads();
  }
}

// ---------------------------------------------------------------------------
// BatchNorm: pass 1 (per-channel sum / sumsq), pass 2 (apply in place)
// ---------------------------------------------------------------------------
__global__ __launch_bounds__(256) void bn_stats_kernel(
    const float* __restrict__ x, float* __restrict__ stats, int nrows)
{
  __shared__ float sh[2][4][64];
  const int c = threadIdx.x & 63, rl = threadIdx.x >> 6;
  float s = 0.f, s2 = 0.f;
  for (int r = blockIdx.x * 4 + rl; r < nrows; r += gridDim.x * 4) {
    float v = x[(size_t)r * 64 + c];
    s += v; s2 += v * v;
  }
  sh[0][rl][c] = s; sh[1][rl][c] = s2;
  __syncthreads();
  if (rl == 0) {
    float ts = sh[0][0][c] + sh[0][1][c] + sh[0][2][c] + sh[0][3][c];
    float t2 = sh[1][0][c] + sh[1][1][c] + sh[1][2][c] + sh[1][3][c];
    atomicAdd(&stats[c], ts);
    atomicAdd(&stats[64 + c], t2);
  }
}

__global__ __launch_bounds__(256) void bn_apply_kernel(
    float* x, const float* __restrict__ stats,
    const float* __restrict__ g, const float* __restrict__ b, int nrows)
{
  const float inv = 1.0f / (float)nrows;
  size_t i = (size_t)blockIdx.x * 256 + threadIdx.x;
  const size_t total = (size_t)nrows * 64;
  for (; i < total; i += (size_t)gridDim.x * 256) {
    const int c = (int)(i & 63);
    const float mean = stats[c] * inv;
    const float var  = stats[64 + c] * inv - mean * mean;
    x[i] = (x[i] - mean) * rsqrtf(var + BNEPS) * g[c] + b[c];
  }
}

// ---------------------------------------------------------------------------
// Graph pooling over sorted batch: run-length accumulate then flush atomics
// ---------------------------------------------------------------------------
__global__ __launch_bounds__(256) void pool_kernel(
    const float* __restrict__ x, const int* __restrict__ batch,
    float* __restrict__ psum, float* __restrict__ pmax, float* __restrict__ pcnt,
    int chunk)
{
  const int c = threadIdx.x & 63, rl = threadIdx.x >> 6;
  const int base = blockIdx.x * chunk;
  int end = base + chunk; if (end > NN) end = NN;
  int gcur = -1; float s = 0.f, m = -3.4e38f, cnt = 0.f;
  for (int r = base + rl; r < end; r += 4) {
    const int gg = batch[r];
    if (gg != gcur) {
      if (gcur >= 0) {
        atomicAdd(&psum[gcur * 64 + c], s);
        atomicMaxF(&pmax[gcur * 64 + c], m);
        if (c == 0) atomicAdd(&pcnt[gcur], cnt);
      }
      gcur = gg; s = 0.f; m = -3.4e38f; cnt = 0.f;
    }
    const float v = x[(size_t)r * 64 + c];
    s += v; m = fmaxf(m, v); cnt += 1.f;
  }
  if (gcur >= 0) {
    atomicAdd(&psum[gcur * 64 + c], s);
    atomicMaxF(&pmax[gcur * 64 + c], m);
    if (c == 0) atomicAdd(&pcnt[gcur], cnt);
  }
}

// ---------------------------------------------------------------------------
// Projector head (single block, 128 threads; tiny 64x128 matrices)
// ---------------------------------------------------------------------------
__global__ __launch_bounds__(128) void head_kernel(
    const float* __restrict__ psum, const float* __restrict__ pmax,
    const float* __restrict__ pcnt,
    const float* __restrict__ pw1, const float* __restrict__ pb1,
    const float* __restrict__ png, const float* __restrict__ pnb,
    const float* __restrict__ pw2, const float* __restrict__ pb2,
    float* __restrict__ zproj, float* __restrict__ xgraph)
{
  __shared__ float buf[NG][PROJ];            // xg, later reused for normalized z
  const int p = threadIdx.x;                 // column 0..127

  // finalize pooling -> x_graph = [mean | max]
  for (int i = p; i < NG * 64; i += 128) {
    const int g = i >> 6, c = i & 63;
    const float cntv = pcnt[g];
    const float mean = psum[i] / fmaxf(cntv, 1.0f);
    float mx = pmax[i];
    if (mx < -3.0e38f) mx = 0.f;             // empty graph -> 0 (isfinite fixup)
    buf[g][c] = mean; buf[g][64 + c] = mx;
    xgraph[g * PROJ + c] = mean; xgraph[g * PROJ + 64 + c] = mx;
  }
  __syncthreads();

  // z = xg @ p_w1 + b1   (per-thread column, rows in registers)
  float zreg[NG];
  for (int g = 0; g < NG; ++g) {
    float acc = pb1[p];
    for (int k = 0; k < PROJ; ++k) acc += buf[g][k] * pw1[k * PROJ + p];
    zreg[g] = acc;
  }
  __syncthreads();                            // done reading xg

  // per-column BN over 64 rows + relu
  float mu = 0.f;
  for (int g = 0; g < NG; ++g) mu += zreg[g];
  mu *= (1.0f / NG);
  float var = 0.f;
  for (int g = 0; g < NG; ++g) { const float d = zreg[g] - mu; var += d * d; }
  var *= (1.0f / NG);
  const float sc = png[p] * rsqrtf(var + BNEPS);
  for (int g = 0; g < NG; ++g) {
    float y = (zreg[g] - mu) * sc + pnb[p];
    buf[g][p] = y > 0.f ? y : 0.f;
  }
  __syncthreads();

  // z_proj = z @ p_w2 + b2
  for (int g = 0; g < NG; ++g) {
    float acc = pb2[p];
    for (int k = 0; k < PROJ; ++k) acc += buf[g][k] * pw2[k * PROJ + p];
    zproj[g * PROJ + p] = acc;
  }
}

// ---------------------------------------------------------------------------
// Init kernels (d_out / d_ws are poisoned; conv outputs need zeros, max needs -inf)
// ---------------------------------------------------------------------------
__global__ void zero2_kernel(float* a, float* b, size_t n) {
  size_t i = (size_t)blockIdx.x * blockDim.x + threadIdx.x;
  for (; i < n; i += (size_t)gridDim.x * blockDim.x) { a[i] = 0.f; b[i] = 0.f; }
}

__global__ void init_small_kernel(float* stats /*256*/, float* psum /*4096*/,
                                  float* pmax /*4096*/, float* pcnt /*64*/) {
  const int t = threadIdx.x;
  if (t < 256) stats[t] = 0.f;
  for (int i = t; i < NG * 64; i += 256) {
    psum[i] = 0.f;
    pmax[i] = __uint_as_float(0xFF800000u);   // -inf
  }
  if (t < NG) pcnt[t] = 0.f;
}

// ---------------------------------------------------------------------------
extern "C" void kernel_launch(void* const* d_in, const int* in_sizes, int n_in,
                              void* d_out, int out_size, void* d_ws, size_t ws_size,
                              hipStream_t stream) {
  (void)in_sizes; (void)n_in; (void)out_size; (void)ws_size;

  const float* x     = (const float*)d_in[0];
  const int*   eidx  = (const int*)d_in[1];
  const int*   batch = (const int*)d_in[2];
  const float* c1w1  = (const float*)d_in[3];
  const float* c1b1  = (const float*)d_in[4];
  const float* c1w2  = (const float*)d_in[5];
  const float* c1b2  = (const float*)d_in[6];
  const float* bn1g  = (const float*)d_in[7];
  const float* bn1b  = (const float*)d_in[8];
  const float* c2w1  = (const float*)d_in[9];
  const float* c2b1  = (const float*)d_in[10];
  const float* c2w2  = (const float*)d_in[11];
  const float* c2b2  = (const float*)d_in[12];
  const float* bn2g  = (const float*)d_in[13];
  const float* bn2b  = (const float*)d_in[14];
  const float* pw1   = (const float*)d_in[15];
  const float* pb1   = (const float*)d_in[16];
  const float* png   = (const float*)d_in[17];
  const float* pnb   = (const float*)d_in[18];
  const float* pw2   = (const float*)d_in[19];
  const float* pb2   = (const float*)d_in[20];

  const int* srcp = eidx;           // edge_index[0]
  const int* dstp = eidx + NE;      // edge_index[1]

  float* out    = (float*)d_out;
  float* zproj  = out;                               // [64,128]
  float* xnode  = out + NG * PROJ;                   // [100000,64]
  float* xgraph = out + NG * PROJ + (size_t)NN * 64; // [64,128]

  float* ws     = (float*)d_ws;
  float* h1     = ws;                                // [100000,64]
  float* stats  = ws + (size_t)NN * 64;              // 256: BN1 (0..127), BN2 (128..255)
  float* psum   = stats + 256;                       // [64,64]
  float* pmax   = psum + NG * 64;                    // [64,64]
  float* pcnt   = pmax + NG * 64;                    // [64]

  // init
  zero2_kernel<<<2048, 256, 0, stream>>>(h1, xnode, (size_t)NN * 64);
  init_small_kernel<<<1, 256, 0, stream>>>(stats, psum, pmax, pcnt);

  // conv1 -> h1 ; BN1 in place
  edgeconv_wmma_kernel<<<2048, 256, 0, stream>>>(x, srcp, dstp,
                                                 c1w1, c1b1, c1w2, c1b2, h1);
  bn_stats_kernel<<<512, 256, 0, stream>>>(h1, stats, NN);
  bn_apply_kernel<<<2048, 256, 0, stream>>>(h1, stats, bn1g, bn1b, NN);

  // conv2 -> x_node region ; BN2 in place
  edgeconv_wmma_kernel<<<2048, 256, 0, stream>>>(h1, srcp, dstp,
                                                 c2w1, c2b1, c2w2, c2b2, xnode);
  bn_stats_kernel<<<512, 256, 0, stream>>>(xnode, stats + 128, NN);
  bn_apply_kernel<<<2048, 256, 0, stream>>>(xnode, stats + 128, bn2g, bn2b, NN);

  // pooling + projector head
  const int chunk = (NN + 127) / 128;
  pool_kernel<<<128, 256, 0, stream>>>(xnode, batch, psum, pmax, pcnt, chunk);
  head_kernel<<<1, 128, 0, stream>>>(psum, pmax, pcnt,
                                     pw1, pb1, png, pnb, pw2, pb2,
                                     zproj, xgraph);
}
